// ScaledDotProductAttention_44702019617051
// MI455X (gfx1250) — compile-verified, matrix operand order checked by reading, not run on hardware
//
#include <hip/hip_runtime.h>
#include <hip/hip_bf16.h>

typedef __attribute__((ext_vector_type(16))) __bf16 v16bf;
typedef __attribute__((ext_vector_type(8)))  float  v8f;
typedef __attribute__((ext_vector_type(4)))  unsigned int v4u;
typedef __attribute__((ext_vector_type(8)))  int v8i;
typedef __attribute__((ext_vector_type(4)))  int v4i;

#define B_  16
#define H_  8
#define L_  1024
#define D_  64
#define SCALE 0.125f   // 1/sqrt(64)

// ---- LDS layout (byte offsets, total 211200 B ~= 206.25 KB, < 320 KB) ----
// sQ   [16][64]  bf16 :      0 ..   2048
// sW   [16][128] bf16 :   2048 ..   6144
// ph1  region (65536 B): phase1 = sKc[128][64] f32 (@6144) + sQc[128][64] f32 (@38912)
//                        phase2/3 = sP[16][1024] bf16 (overlay @6144)
// sS1  [16][1024] f32 :  71680 .. 137216
// sS2  [16][1024] f32 : 137216 .. 202752
// sCtx [2][16][64] f32: 202752 .. 210944
// sStat[16][4] f32    : 210944 .. 211200
#define OFF_SQ    0
#define OFF_SW    2048
#define OFF_KC    6144
#define OFF_QC    38912
#define OFF_SP    6144
#define OFF_S1    71680
#define OFF_S2    137216
#define OFF_CTX   202752
#define OFF_STAT  210944
#define SMEM_BYTES 211200

__device__ __forceinline__ v8f wmma_bf16(v16bf a, v16bf b, v8f c) {
    return __builtin_amdgcn_wmma_f32_16x16x32_bf16(
        false, a, false, b, (short)0, c, false, false);
}

// 16x16x32 bf16 fragment from row-major bf16 tile (lane = row/col, hlf = lane>>4)
__device__ __forceinline__ v16bf load_frag(const __bf16* p, int hlf) {
    v16bf f;
#pragma unroll
    for (int j = 0; j < 16; ++j) {
        int c = ((j & 7) | ((j & 8) << 1)) + hlf * 8;
        f[j] = p[c];
    }
    return f;
}

// Same fragment pattern but from an f32 tile (TDM-staged), converting to bf16
__device__ __forceinline__ v16bf load_frag_f32(const float* p, int hlf) {
    v16bf f;
#pragma unroll
    for (int j = 0; j < 16; ++j) {
        int c = ((j & 7) | ((j & 8) << 1)) + hlf * 8;
        f[j] = (__bf16)p[c];
    }
    return f;
}

// Issue a TDM 2D tile load: rows x rowlen f32 elements, contiguous rows,
// global -> LDS at byte offset lds_off. D# packed per CDNA5 ISA §8.3/§8.4.
// This toolchain exposes the 6-arg builtin: (g0, g1, g2, g3, g4, cpol).
__device__ __forceinline__ void tdm_load_2d(unsigned int lds_off, const void* gptr,
                                            unsigned int rows, unsigned int rowlen) {
    unsigned long long ga = (unsigned long long)(uintptr_t)gptr;
    v4u g0;
    g0[0] = 1u;                                              // count=1, user mode
    g0[1] = lds_off;                                         // lds_addr (bytes)
    g0[2] = (unsigned int)ga;                                // global_addr[31:0]
    g0[3] = (unsigned int)((ga >> 32) & 0x01FFFFFFu)         // global_addr[56:32]
          | (2u << 30);                                      // type = 2 ("image")
    v8i g1;
    g1[0] = (int)(2u << 16);                                 // data_size = 4B
    g1[1] = (int)(rowlen << 16);                             // tensor_dim0[15:0]
    g1[2] = (int)(((rows & 0xFFFFu) << 16) | (rowlen >> 16));// tensor_dim1 | dim0 hi
    g1[3] = (int)(((rowlen & 0xFFFFu) << 16) | (rows >> 16));// tile_dim0 | dim1 hi
    g1[4] = (int)(rows & 0xFFFFu);                           // tile_dim1 (tile_dim2=0)
    g1[5] = (int)rowlen;                                     // tensor_dim0_stride lo
    g1[6] = 0;                                               // d0_stride hi | d1_stride lo
    g1[7] = 0;                                               // d1_stride hi
    v4i g2 = {0, 0, 0, 0};
    v4i g3 = {0, 0, 0, 0};
    v8i g4 = {0, 0, 0, 0, 0, 0, 0, 0};
    __builtin_amdgcn_tensor_load_to_lds(g0, g1, g2, g3, g4, 0);
}

__global__ __launch_bounds__(256)
void dual_attn_kernel(const float* __restrict__ Q, const float* __restrict__ K,
                      const float* __restrict__ V, const float* __restrict__ W,
                      float* __restrict__ ctx_out, float* __restrict__ attn_out)
{
    const int qt   = blockIdx.x;          // query tile 0..63
    const int bh   = blockIdx.y;          // 0..127
    const int h    = bh & (H_ - 1);
    const int tid  = threadIdx.x;
    const int lane = tid & 31;
    const int wave = tid >> 5;            // 0..7
    const int lrow = lane & 15;
    const int hlf  = lane >> 4;
    const int q0   = qt * 16;

    __shared__ __align__(16) char smem[SMEM_BYTES];
    __bf16* sQ   = (__bf16*)(smem + OFF_SQ);
    __bf16* sW   = (__bf16*)(smem + OFF_SW);
    float*  sKc  = (float*)(smem + OFF_KC);     // phase 1 (TDM dest)
    float*  sQc  = (float*)(smem + OFF_QC);     // phase 1 (TDM dest)
    __bf16* sP   = (__bf16*)(smem + OFF_SP);    // phase 2/3 overlay
    float*  sS1  = (float*)(smem + OFF_S1);
    float*  sS2  = (float*)(smem + OFF_S2);
    float*  sCtx = (float*)(smem + OFF_CTX);
    float*  sStat= (float*)(smem + OFF_STAT);

    const float* Qbh = Q + (size_t)bh * L_ * D_;
    const float* Kbh = K + (size_t)bh * L_ * D_;
    const float* Vbh = V + (size_t)bh * L_ * D_;
    const float* Wh  = W + (size_t)h  * L_ * (2 * D_);

    // ---- Phase 0: stage Q query-tile and W rows as bf16 ----
    for (int i = tid; i < 16 * 64; i += 256)
        sQ[i] = (__bf16)Qbh[(size_t)(q0 + (i >> 6)) * D_ + (i & 63)];
    for (int i = tid; i < 16 * 128; i += 256)
        sW[i] = (__bf16)Wh[(size_t)(q0 + (i >> 7)) * (2 * D_) + (i & 127)];
    __syncthreads();

    // Loop-invariant A fragments (query rows / W rows)
    v16bf aQ0 = load_frag(sQ + lrow * 64, hlf);
    v16bf aQ1 = load_frag(sQ + lrow * 64 + 32, hlf);
    v16bf aW0 = load_frag(sW + lrow * 128, hlf);
    v16bf aW1 = load_frag(sW + lrow * 128 + 32, hlf);
    v16bf aW2 = load_frag(sW + lrow * 128 + 64, hlf);
    v16bf aW3 = load_frag(sW + lrow * 128 + 96, hlf);

    // ---- Phase 1: scores for all 1024 keys, key chunks of 128 ----
    for (int kc = 0; kc < 8; ++kc) {
        const int kbase = kc * 128;

        // TDM: DMA K/Q chunks (128 rows x 64 f32) straight into LDS
        if (wave == 0) {
            tdm_load_2d(OFF_KC, Kbh + (size_t)kbase * D_, 128u, 64u);
            tdm_load_2d(OFF_QC, Qbh + (size_t)kbase * D_, 128u, 64u);
            __builtin_amdgcn_s_wait_tensorcnt(0);
        }
        // stream V into cache for phase 3 (strided gather later)
        __builtin_prefetch(Vbh + (size_t)kbase * D_ + tid * 32, 0, 0);
        __syncthreads();

        // this wave's 16-key tile
        const float* kp = sKc + (wave * 16 + lrow) * 64;
        const float* qp = sQc + (wave * 16 + lrow) * 64;
        v16bf bK0 = load_frag_f32(kp, hlf),  bK1 = load_frag_f32(kp + 32, hlf);
        v16bf bQ0 = load_frag_f32(qp, hlf),  bQ1 = load_frag_f32(qp + 32, hlf);

        v8f z = {};
        v8f s1 = wmma_bf16(aQ0, bK0, z);
        s1     = wmma_bf16(aQ1, bK1, s1);
        v8f s2 = wmma_bf16(aW0, bQ0, z);
        s2     = wmma_bf16(aW1, bQ1, s2);
        s2     = wmma_bf16(aW2, bK0, s2);
        s2     = wmma_bf16(aW3, bK1, s2);

        const int keycol = kbase + wave * 16 + lrow;   // D: N = lane&15
#pragma unroll
        for (int r = 0; r < 8; ++r) {                  // D: M = r + 8*(lane>>4)
            int row = r + hlf * 8;
            sS1[row * 1024 + keycol] = s1[r] * SCALE;
            sS2[row * 1024 + keycol] = s2[r] * SCALE;
        }
        __syncthreads();
    }

    // ---- Phase 2a: per-row softmax stats (2 rows per wave) ----
    for (int rr = 0; rr < 2; ++rr) {
        const int row = wave * 2 + rr;
        const float* p1 = sS1 + row * 1024;
        const float* p2 = sS2 + row * 1024;
        float m1 = -3.0e38f, m2 = -3.0e38f;
        for (int c = lane; c < 1024; c += 32) {
            m1 = fmaxf(m1, p1[c]);  m2 = fmaxf(m2, p2[c]);
        }
#pragma unroll
        for (int o = 16; o; o >>= 1) {
            m1 = fmaxf(m1, __shfl_xor(m1, o, 32));
            m2 = fmaxf(m2, __shfl_xor(m2, o, 32));
        }
        float t1 = 0.f, t2 = 0.f;
        for (int c = lane; c < 1024; c += 32) {
            t1 += __expf(p1[c] - m1);  t2 += __expf(p2[c] - m2);
        }
#pragma unroll
        for (int o = 16; o; o >>= 1) {
            t1 += __shfl_xor(t1, o, 32);
            t2 += __shfl_xor(t2, o, 32);
        }
        if (lane == 0) {
            sStat[row * 4 + 0] = m1;  sStat[row * 4 + 1] = 1.f / t1;
            sStat[row * 4 + 2] = m2;  sStat[row * 4 + 3] = 1.f / t2;
        }
    }
    __syncthreads();

    // ---- Phase 2b: p = softmax1 + softmax2 -> global f32 + LDS bf16 ----
    for (int i = tid; i < 16 * 1024; i += 256) {
        const int row = i >> 10, col = i & 1023;
        float p = __expf(sS1[i] - sStat[row * 4 + 0]) * sStat[row * 4 + 1]
                + __expf(sS2[i] - sStat[row * 4 + 2]) * sStat[row * 4 + 3];
        attn_out[((size_t)(bh * L_ + q0 + row)) * L_ + col] = p;
        sP[i] = (__bf16)p;
    }
    __syncthreads();

    // ---- Phase 3: context = P[16x1024] x V[1024x64] ----
    const int chalf = wave >> 2;    // key half (0: keys 0-511, 1: 512-1023)
    const int colt  = wave & 3;     // 16-col tile of V
    v8f acc = {};
    for (int i = 0; i < 16; ++i) {
        const int key0 = chalf * 512 + i * 32;
        v16bf a = load_frag(sP + lrow * 1024 + key0, hlf);
        v16bf b;
#pragma unroll
        for (int j = 0; j < 16; ++j) {
            int cc = ((j & 7) | ((j & 8) << 1)) + hlf * 8;
            b[j] = (__bf16)Vbh[(size_t)(key0 + cc) * D_ + colt * 16 + lrow];
        }
        acc = wmma_bf16(a, b, acc);
    }
#pragma unroll
    for (int r = 0; r < 8; ++r)
        sCtx[(chalf * 16 + r + hlf * 8) * 64 + colt * 16 + lrow] = acc[r];
    __syncthreads();

    for (int i = tid; i < 16 * 64; i += 256) {
        const int row = i >> 6, col = i & 63;
        ctx_out[((size_t)(bh * L_ + q0 + row)) * D_ + col] =
            sCtx[row * 64 + col] + sCtx[1024 + row * 64 + col];
    }
}

extern "C" void kernel_launch(void* const* d_in, const int* in_sizes, int n_in,
                              void* d_out, int out_size, void* d_ws, size_t ws_size,
                              hipStream_t stream) {
    const float* Q = (const float*)d_in[0];
    const float* K = (const float*)d_in[1];
    const float* V = (const float*)d_in[2];
    const float* W = (const float*)d_in[3];
    float* ctx  = (float*)d_out;                               // [B,H,L,64]
    float* attn = ctx + (size_t)B_ * H_ * L_ * D_;             // [B,H,L,L]
    dim3 grid(L_ / 16, B_ * H_);
    dim3 block(256);
    dual_attn_kernel<<<grid, block, 0, stream>>>(Q, K, V, W, ctx, attn);
}
